// AttnBlock3D_74792560492866
// MI455X (gfx1250) — matrix operand order repaired and checked by reading, compiled, not run on hardware
//
#include <hip/hip_runtime.h>

#define C_ 256
#define N_ 4096
#define B_ 2
#define G_ 32

typedef _Float16 half8 __attribute__((ext_vector_type(8)));
typedef _Float16 v16h  __attribute__((ext_vector_type(16)));
typedef float    v8f   __attribute__((ext_vector_type(8)));
typedef unsigned int u32x4 __attribute__((ext_vector_type(4)));
typedef int      i32x8 __attribute__((ext_vector_type(8)));
typedef int      i32x4 __attribute__((ext_vector_type(4)));

union V16U { v16h v; half8 g[2]; };
union F8U  { v8f  v; float f[8]; };

#define WMMA16(A, Bm, Cv) \
  __builtin_amdgcn_wmma_f32_16x16x32_f16(false, (A), false, (Bm), (short)0, (Cv), false, false)

// A-operand fragment (16x32 f16): row = row0 + lane%16, k halves at
// [k0 + hi*8 .. +7] and [k0 + 16 + hi*8 .. +7]  (ISA 7.12.2).
__device__ __forceinline__ v16h load_fragA(const _Float16* base, int row0, int rstride, int k0) {
  int lane = threadIdx.x & 31;
  const _Float16* p = base + (size_t)(row0 + (lane & 15)) * rstride + k0 + ((lane >> 4) << 3);
  V16U u;
  u.g[0] = *(const half8*)p;
  u.g[1] = *(const half8*)(p + 16);
  return u.v;
}

// B-operand fragment (32x16 f16): column = row0 + lane%16 (storage row),
// k halves at [k0 + hi*8 .. +7] and [k0 + 16 + hi*8 .. +7].
__device__ __forceinline__ v16h load_fragB(const _Float16* base, int row0, int rstride, int k0) {
  int lane = threadIdx.x & 31;
  const _Float16* p = base + (size_t)(row0 + (lane & 15)) * rstride + k0 + ((lane >> 4) << 3);
  V16U u;
  u.g[0] = *(const half8*)p;
  u.g[1] = *(const half8*)(p + 16);
  return u.v;
}

// A-fragment from padded LDS staging buffer.
__device__ __forceinline__ v16h load_fragA_lds(const _Float16* base, int rstride) {
  int lane = threadIdx.x & 31;
  const _Float16* p = base + (lane & 15) * rstride + ((lane >> 4) << 3);
  V16U u;
  u.g[0] = *(const half8*)p;
  u.g[1] = *(const half8*)(p + 16);
  return u.v;
}

// B-fragment from padded LDS tile (V tile staged by TDM, row stride in halves).
__device__ __forceinline__ v16h load_fragB_lds(const _Float16* base, int row0, int rstride) {
  int lane = threadIdx.x & 31;
  const _Float16* p = base + (row0 + (lane & 15)) * rstride + ((lane >> 4) << 3);
  V16U u;
  u.g[0] = *(const half8*)p;
  u.g[1] = *(const half8*)(p + 16);
  return u.v;
}

// TDM: DMA one V tile (256 rows x 64B, row stride 8192B) from global into LDS,
// padded 4 DWORDs every 16 DWORDs -> LDS row stride 80B (16B-aligned, bank-conflict-free).
// D# per CDNA5 ISA sec. 8.3/8.4; data_size=8B units. 6-arg builtin (clang-23 toolchain).
__device__ __forceinline__ void tdm_load_v_tile(const _Float16* gsrc, unsigned int lds_off) {
  unsigned long long ga = (unsigned long long)(uintptr_t)gsrc;
  u32x4 g0 = { 1u,                                   // count=1, user descriptor
               lds_off,                               // lds_addr (bytes)
               (unsigned int)ga,                      // global_addr[31:0]
               ((unsigned int)(ga >> 32) & 0x1FFFFFFu) | 0x80000000u }; // [56:32] | type=2
  i32x8 g1 = { (int)0x06D30000,     // data_size=3(8B), pad_enable, pad_interval=3(16dw), pad_amount=3(4dw)
               (int)(1024u << 16),  // tensor_dim0[15:0]=1024 units (8192B row)
               (int)(256u << 16),   // tensor_dim0 hi=0 | tensor_dim1[15:0]=256
               (int)(8u << 16),     // tensor_dim1 hi=0 | tile_dim0=8 units (64B)
               256,                 // tile_dim1=256 rows, tile_dim2=0
               1024,                // tensor_dim0_stride[31:0]=1024 units
               0, 0 };              // stride hi / dim1_stride unused (2D)
  i32x4 z4 = { 0, 0, 0, 0 };
  i32x8 z8 = { 0, 0, 0, 0, 0, 0, 0, 0 };
  __builtin_amdgcn_tensor_load_to_lds(g0, g1, z4, z4, z8, 0);
}

// ---------------- GroupNorm (fused stats + apply + (C,N)->(N,C) f16 transpose) ----
__global__ __launch_bounds__(256) void gn_kernel(const float* __restrict__ x,
                                                 const float* __restrict__ gw,
                                                 const float* __restrict__ gb,
                                                 _Float16* __restrict__ h_t) {
  __shared__ float s_sum[256];
  __shared__ float s_sq[256];
  int b = blockIdx.x / G_;
  int g = blockIdx.x % G_;
  int c0 = g * (C_ / G_);
  const float* base = x + ((size_t)b * C_ + c0) * N_;
  const int M = (C_ / G_) * N_;                     // 32768
  int tid = threadIdx.x;
  float s = 0.f, s2 = 0.f;
  for (int i = tid; i < M; i += 256) {
    float v = base[i];
    s += v; s2 += v * v;
  }
  s_sum[tid] = s; s_sq[tid] = s2;
  __syncthreads();
  for (int off = 128; off > 0; off >>= 1) {
    if (tid < off) { s_sum[tid] += s_sum[tid + off]; s_sq[tid] += s_sq[tid + off]; }
    __syncthreads();
  }
  float mean = s_sum[0] * (1.0f / M);
  float var  = s_sq[0] * (1.0f / M) - mean * mean;
  float rstd = rsqrtf(var + 1e-5f);
  _Float16* ho = h_t + (size_t)b * N_ * C_;
  for (int i = tid; i < M; i += 256) {
    int c = c0 + (i >> 12);
    int n = i & (N_ - 1);
    float v = (base[i] - mean) * rstd * gw[c] + gb[c];
    ho[(size_t)n * C_ + c] = (_Float16)v;
  }
}

// ---------------- weight f32 -> f16 ------------------------------------------------
__global__ __launch_bounds__(256) void wcvt_kernel(const float* __restrict__ wq,
                                                   const float* __restrict__ wk,
                                                   const float* __restrict__ wv,
                                                   const float* __restrict__ wp,
                                                   _Float16* __restrict__ w16) {
  int i = blockIdx.x * 256 + threadIdx.x;
  w16[i]          = (_Float16)wq[i];
  w16[65536 + i]  = (_Float16)wk[i];
  w16[131072 + i] = (_Float16)wv[i];
  w16[196608 + i] = (_Float16)wp[i];
}

// -------- Q/K projection, 4 n-tiles per wave sharing the weight fragment ----------
__global__ __launch_bounds__(256) void gemm_qk_kernel(const _Float16* __restrict__ h_t,
                                                      const _Float16* __restrict__ w16q,
                                                      const _Float16* __restrict__ w16k,
                                                      const float* __restrict__ bq,
                                                      const float* __restrict__ bk,
                                                      _Float16* __restrict__ q_t,
                                                      _Float16* __restrict__ k_t) {
  int b = blockIdx.y;
  int is_k = blockIdx.z;
  const _Float16* w = is_k ? w16k : w16q;
  const float* bias = is_k ? bk : bq;
  _Float16* out     = (is_k ? k_t : q_t) + (size_t)b * N_ * C_;
  const _Float16* h = h_t + (size_t)b * N_ * C_;
  int wave = threadIdx.x >> 5, lane = threadIdx.x & 31;
  int task = blockIdx.x * 8 + wave;                 // 1024 tasks: 16 o x 64 n-groups
  int o0 = (task & 15) * 16;
  int n0 = (task >> 4) * 64;
  F8U acc[4];
#pragma unroll
  for (int m = 0; m < 4; ++m)
#pragma unroll
    for (int r = 0; r < 8; ++r) acc[m].f[r] = 0.f;
#pragma unroll
  for (int k = 0; k < C_; k += 32) {
    v16h wb = load_fragB(w, o0, C_, k);
#pragma unroll
    for (int m = 0; m < 4; ++m) {
      v16h a = load_fragA(h, n0 + m * 16, C_, k);
      acc[m].v = WMMA16(a, wb, acc[m].v);
    }
  }
  int hi = lane >> 4, ln = lane & 15;
  float bo = bias[o0 + ln];
#pragma unroll
  for (int m = 0; m < 4; ++m)
#pragma unroll
    for (int r = 0; r < 8; ++r) {
      int n = n0 + m * 16 + r + hi * 8;
      out[(size_t)n * C_ + o0 + ln] = (_Float16)(acc[m].f[r] + bo);
    }
}

// -------- V projection, 4 m-tiles per wave sharing the weight fragment, (C,N) -----
__global__ __launch_bounds__(256) void gemm_v_kernel(const _Float16* __restrict__ h_t,
                                                     const _Float16* __restrict__ w16v,
                                                     const float* __restrict__ bv,
                                                     _Float16* __restrict__ vout) {
  int b = blockIdx.y;
  const _Float16* h = h_t + (size_t)b * N_ * C_;
  _Float16* out = vout + (size_t)b * C_ * N_;
  int wave = threadIdx.x >> 5, lane = threadIdx.x & 31;
  int task = blockIdx.x * 8 + wave;                 // 16 o x 64 m-groups
  int o0 = (task & 15) * 16;
  int m0 = (task >> 4) * 64;
  F8U acc[4];
#pragma unroll
  for (int m = 0; m < 4; ++m)
#pragma unroll
    for (int r = 0; r < 8; ++r) acc[m].f[r] = 0.f;
#pragma unroll
  for (int k = 0; k < C_; k += 32) {
    v16h a = load_fragA(w16v, o0, C_, k);
#pragma unroll
    for (int m = 0; m < 4; ++m) {
      v16h bb = load_fragB(h, m0 + m * 16, C_, k);
      acc[m].v = WMMA16(a, bb, acc[m].v);
    }
  }
  int hi = lane >> 4, ln = lane & 15;
#pragma unroll
  for (int m = 0; m < 4; ++m)
#pragma unroll
    for (int r = 0; r < 8; ++r) {
      int o = o0 + r + hi * 8;
      out[(size_t)o * N_ + m0 + m * 16 + ln] = (_Float16)(acc[m].f[r] + bv[o]);
    }
}

// ------- Flash attention + output projection + residual; V staged via TDM ---------
#define VROW 40                                     // LDS V row stride in halves (80B)
__global__ __launch_bounds__(256) void attn_kernel(const _Float16* __restrict__ q_t,
                                                   const _Float16* __restrict__ k_t,
                                                   const _Float16* __restrict__ v,
                                                   const _Float16* __restrict__ wp16,
                                                   const float* __restrict__ bp,
                                                   const float* __restrict__ x,
                                                   float* __restrict__ out) {
  __shared__ __align__(16) _Float16 pstage[8][16][40];  // per-wave P/O transpose staging
  __shared__ __align__(16) _Float16 vstage[2][C_ * VROW]; // double-buffered TDM V tile
  int b = blockIdx.y;
  int wave = threadIdx.x >> 5, lane = threadIdx.x & 31;
  int hi = lane >> 4, ln = lane & 15;
  int n0 = blockIdx.x * 128 + wave * 16;

  const _Float16* q  = q_t + (size_t)b * N_ * C_;
  const _Float16* kk = k_t + (size_t)b * N_ * C_;
  const _Float16* vv = v   + (size_t)b * C_ * N_;
  _Float16 (*pl)[40] = pstage[wave];

  v16h qf[8];
#pragma unroll
  for (int k = 0; k < 8; ++k) qf[k] = load_fragA(q, n0, C_, k * 32);

  F8U oac[16];
#pragma unroll
  for (int t = 0; t < 16; ++t)
#pragma unroll
    for (int r = 0; r < 8; ++r) oac[t].f[r] = 0.f;

  float mrow[8], lrow[8];
#pragma unroll
  for (int r = 0; r < 8; ++r) { mrow[r] = -3.0e38f; lrow[r] = 0.f; }

  const float scale = 0.0625f;                      // 256^-0.5
  unsigned int voff[2];
  voff[0] = (unsigned int)(uintptr_t)(&vstage[0][0]);
  voff[1] = (unsigned int)(uintptr_t)(&vstage[1][0]);

  if (wave == 0) tdm_load_v_tile(vv, voff[0]);      // prologue DMA

  const int NIT = N_ / 32;
  for (int it = 0; it < NIT; ++it) {
    int j0 = it << 5;
    int cur = it & 1;
    if (wave == 0) __builtin_amdgcn_s_wait_tensorcnt(0);
    __syncthreads();                                // V tile `cur` visible to all waves
    if (wave == 0 && it + 1 < NIT)
      tdm_load_v_tile(vv + j0 + 32, voff[cur ^ 1]); // overlap next DMA with compute
    const _Float16* vl = &vstage[cur][0];

    // ---- S = Q^T K for 32 keys ----
    F8U s0, s1;
#pragma unroll
    for (int r = 0; r < 8; ++r) { s0.f[r] = 0.f; s1.f[r] = 0.f; }
#pragma unroll
    for (int k = 0; k < 8; ++k) {
      v16h b0 = load_fragB(kk, j0,      C_, k * 32);
      v16h b1 = load_fragB(kk, j0 + 16, C_, k * 32);
      s0.v = WMMA16(qf[k], b0, s0.v);
      s1.v = WMMA16(qf[k], b1, s1.v);
    }
    if (j0 + 32 < N_)
      __builtin_prefetch(kk + (size_t)(j0 + 32 + lane) * C_, 0, 0);

    // ---- online softmax over the 32 new columns ----
#pragma unroll
    for (int r = 0; r < 8; ++r) {
      float a0 = s0.f[r] * scale, a1 = s1.f[r] * scale;
      float mx = fmaxf(a0, a1);
#pragma unroll
      for (int d = 1; d < 16; d <<= 1) mx = fmaxf(mx, __shfl_xor(mx, d, 32));
      float mnew = fmaxf(mrow[r], mx);
      float alpha = __expf(mrow[r] - mnew);
      mrow[r] = mnew;
      float p0 = __expf(a0 - mnew);
      float p1 = __expf(a1 - mnew);
      float ps = p0 + p1;
#pragma unroll
      for (int d = 1; d < 16; d <<= 1) ps += __shfl_xor(ps, d, 32);
      lrow[r] = lrow[r] * alpha + ps;
#pragma unroll
      for (int t = 0; t < 16; ++t) oac[t].f[r] *= alpha;
      int n = r + hi * 8;
      pl[n][ln]      = (_Float16)p0;
      pl[n][16 + ln] = (_Float16)p1;
    }
    asm volatile("s_wait_dscnt 0" ::: "memory");
    v16h pf = load_fragA_lds(&pl[0][0], 40);

    // ---- O += P V^T from the LDS-staged tile ----
#pragma unroll
    for (int t = 0; t < 16; ++t) {
      v16h vb = load_fragB_lds(vl, t * 16, VROW);
      oac[t].v = WMMA16(pf, vb, oac[t].v);
    }
    __syncthreads();                                // done reading `cur` before reuse
  }

  // ---- normalize ----
#pragma unroll
  for (int r = 0; r < 8; ++r) {
    float inv = 1.0f / lrow[r];
#pragma unroll
    for (int t = 0; t < 16; ++t) oac[t].f[r] *= inv;
  }

  // ---- fused output projection: res[n][o] = sum_c O[n][c] wp[o][c] + bp[o] ----
  F8U acc[16];
#pragma unroll
  for (int t = 0; t < 16; ++t) {
    float bo = bp[t * 16 + ln];
#pragma unroll
    for (int r = 0; r < 8; ++r) acc[t].f[r] = bo;
  }
#pragma unroll
  for (int k = 0; k < 8; ++k) {
#pragma unroll
    for (int r = 0; r < 8; ++r) {
      int n = r + hi * 8;
      pl[n][ln]      = (_Float16)oac[2 * k].f[r];
      pl[n][16 + ln] = (_Float16)oac[2 * k + 1].f[r];
    }
    asm volatile("s_wait_dscnt 0" ::: "memory");
    v16h of = load_fragA_lds(&pl[0][0], 40);
#pragma unroll
    for (int t = 0; t < 16; ++t) {
      v16h wf = load_fragB(wp16, t * 16, C_, k * 32);
      acc[t].v = WMMA16(of, wf, acc[t].v);
    }
  }

  // ---- residual add + store ----
#pragma unroll
  for (int t = 0; t < 16; ++t) {
    int och = t * 16 + ln;
    size_t base = ((size_t)b * C_ + och) * N_ + n0 + hi * 8;
    const float4* xp = (const float4*)(x + base);
    float4 x0 = xp[0], x1 = xp[1];
    float4 y0, y1;
    y0.x = x0.x + acc[t].f[0]; y0.y = x0.y + acc[t].f[1];
    y0.z = x0.z + acc[t].f[2]; y0.w = x0.w + acc[t].f[3];
    y1.x = x1.x + acc[t].f[4]; y1.y = x1.y + acc[t].f[5];
    y1.z = x1.z + acc[t].f[6]; y1.w = x1.w + acc[t].f[7];
    ((float4*)(out + base))[0] = y0;
    ((float4*)(out + base))[1] = y1;
  }
}

extern "C" void kernel_launch(void* const* d_in, const int* in_sizes, int n_in,
                              void* d_out, int out_size, void* d_ws, size_t ws_size,
                              hipStream_t stream) {
  const float* x   = (const float*)d_in[0];
  const float* gnw = (const float*)d_in[1];
  const float* gnb = (const float*)d_in[2];
  const float* wq  = (const float*)d_in[3];
  const float* bq  = (const float*)d_in[4];
  const float* wk  = (const float*)d_in[5];
  const float* bk  = (const float*)d_in[6];
  const float* wv  = (const float*)d_in[7];
  const float* bv  = (const float*)d_in[8];
  const float* wp  = (const float*)d_in[9];
  const float* bp  = (const float*)d_in[10];
  float* out = (float*)d_out;

  char* ws = (char*)d_ws;
  _Float16* w16  = (_Float16*)ws;                              // 4 x 64K halves
  _Float16* h_t  = (_Float16*)(ws + 524288);                   // (B,N,C) f16
  _Float16* q_t  = (_Float16*)(ws + 524288 + 1 * 4194304);     // (B,N,C) f16
  _Float16* k_t  = (_Float16*)(ws + 524288 + 2 * 4194304);     // (B,N,C) f16
  _Float16* vbuf = (_Float16*)(ws + 524288 + 3 * 4194304);     // (B,C,N) f16

  gn_kernel<<<dim3(B_ * G_), 256, 0, stream>>>(x, gnw, gnb, h_t);
  wcvt_kernel<<<dim3(256), 256, 0, stream>>>(wq, wk, wv, wp, w16);
  gemm_qk_kernel<<<dim3(128, B_, 2), 256, 0, stream>>>(h_t, w16, w16 + 65536, bq, bk, q_t, k_t);
  gemm_v_kernel<<<dim3(128, B_), 256, 0, stream>>>(h_t, w16 + 131072, bv, vbuf);
  attn_kernel<<<dim3(N_ / 128, B_), 256, 0, stream>>>(q_t, k_t, vbuf, w16 + 196608, bp, x, out);
}
